// SelfAttentionPooling_77549929497260
// MI455X (gfx1250) — compile-verified, matrix operand order checked by reading, not run on hardware
//
#include <hip/hip_runtime.h>
#include <hip/hip_bf16.h>
#include <math.h>

typedef __attribute__((ext_vector_type(16))) __bf16   v16bf;
typedef __attribute__((ext_vector_type(8)))  float    v8f;
typedef __attribute__((ext_vector_type(8)))  unsigned v8u;

static constexpr int kB = 16;
static constexpr int kS = 4096;
static constexpr int kH = 1024;
static constexpr int kU = 128;
static constexpr int kA = 16;
static constexpr float kNEG = -1e20f;

// Native CDNA5 V_TANH_F32 if the toolchain exposes it; libm fallback otherwise.
__device__ __forceinline__ float fast_tanh(float x) {
#if __has_builtin(__builtin_amdgcn_tanhf)
    return __builtin_amdgcn_tanhf(x);
#elif __has_builtin(__builtin_amdgcn_tanh_f32)
    return __builtin_amdgcn_tanh_f32(x);
#else
    return tanhf(x);
#endif
}

// cheap scalar fp32 -> bf16 (round-half-up): 1 add + 1 shift
__device__ __forceinline__ __bf16 f2bf(float f) {
    unsigned u = __builtin_bit_cast(unsigned, f) + 0x8000u;
    unsigned short h = (unsigned short)(u >> 16);
    return __builtin_bit_cast(__bf16, h);
}

// pack two fp32 into one dword of two bf16 (a -> [15:0], b -> [31:16])
__device__ __forceinline__ unsigned pack_bf16(float a, float b) {
    unsigned ua = __builtin_bit_cast(unsigned, a) + 0x8000u;
    unsigned ub = __builtin_bit_cast(unsigned, b) + 0x8000u;
#if __has_builtin(__builtin_amdgcn_perm)
    return __builtin_amdgcn_perm(ub, ua, 0x07060302u);   // v_perm_b32
#else
    return (ua >> 16) | (ub & 0xffff0000u);
#endif
}

__device__ __forceinline__ v16bf frag_from_lds(const __bf16* p /*16B aligned*/) {
    const uint4* q = (const uint4*)p;
    uint4 lo = q[0];          // bf16 elements +0..+7
    uint4 hi = q[2];          // bf16 elements +16..+23
    v8u u;
    u[0] = lo.x; u[1] = lo.y; u[2] = lo.z; u[3] = lo.w;
    u[4] = hi.x; u[5] = hi.y; u[6] = hi.z; u[7] = hi.w;
    return __builtin_bit_cast(v16bf, u);
}

// ---------------------------------------------------------------------------
// K0: pre-convert + pre-tile W1 (1024x128 f32) -> bf16 [chunk][col][k]
// w1t[(k/32)*4096 + col*32 + (k%32)] so each 32-K chunk is one contiguous 8 KB
// blob already in the B-fragment LDS order. Runs once, 512 KB of traffic.
// ---------------------------------------------------------------------------
__global__ __launch_bounds__(256) void k0_w1pack(
    const float* __restrict__ W1, __bf16* __restrict__ w1t)
{
    int i = blockIdx.x * 256 + threadIdx.x;     // 0 .. 131071, coalesced read
    int k = i / kU, col = i % kU;
    w1t[(k >> 5) * (kU * 32) + col * 32 + (k & 31)] = f2bf(W1[i]);
}

// ---------------------------------------------------------------------------
// K1: logits[b*s, a] = mask-applied ( tanh(data @ W1) @ W2 )
// 128 rows of flattened (b,s) per block; 8 waves, each owns a 16-row tile.
// ---------------------------------------------------------------------------
__global__ __launch_bounds__(256) void k1_logits(
    const float* __restrict__ data, const float* __restrict__ mask,
    const __bf16* __restrict__ w1t, const float* __restrict__ W2,
    float* __restrict__ logits)
{
    __shared__ __align__(16) __bf16 w1s[kU * 32];      // [col][k], 32-K chunk  (8 KB)
    __shared__ __align__(16) __bf16 w2s[kA * kU];      // [a][k]                (4 KB)
    __shared__ __align__(16) __bf16 hids[8][16 * kU];  // per-wave tanh [m][u]  (32 KB)

    const int tid    = threadIdx.x;
    const int wave   = tid >> 5;
    const int lane   = tid & 31;
    const int lane16 = lane & 15;
    const int hi     = lane >> 4;          // K-half select per ISA A/B layout

    const int rowTile0 = blockIdx.x * 128;
    const int myRow    = rowTile0 + wave * 16 + lane16;

    // stage W2 (U x A, row-major) -> LDS [a][k] as bf16
    for (int i = tid; i < kU * kA; i += 256) {
        int k = i / kA, a = i % kA;
        w2s[a * kU + k] = f2bf(W2[i]);
    }

    v8f zero = {0.f,0.f,0.f,0.f,0.f,0.f,0.f,0.f};
    v8f acc[8];
    #pragma unroll
    for (int j = 0; j < 8; ++j) acc[j] = zero;

    for (int k0 = 0; k0 < kH; k0 += 32) {
        __syncthreads();
        // stage pre-tiled W1 chunk: pure contiguous 8 KB copy (b128 in, b128 out)
        {
            const uint4* src = (const uint4*)(w1t + (k0 >> 5) * (kU * 32));
            uint4* dst = (uint4*)w1s;
            dst[tid]       = src[tid];
            dst[tid + 256] = src[tid + 256];
        }
        __syncthreads();

        // A fragment (16x32 bf16): lane<16 -> K {0..7,16..23}; lane>=16 -> {8..15,24..31}
        const float4* ap = (const float4*)(data + (size_t)myRow * kH + k0 + hi * 8);
        float4 x0 = ap[0], x1 = ap[1];     // k offsets +0..+7
        float4 x2 = ap[4], x3 = ap[5];     // k offsets +16..+23
        v8u au;
        au[0] = pack_bf16(x0.x, x0.y);  au[1] = pack_bf16(x0.z, x0.w);
        au[2] = pack_bf16(x1.x, x1.y);  au[3] = pack_bf16(x1.z, x1.w);
        au[4] = pack_bf16(x2.x, x2.y);  au[5] = pack_bf16(x2.z, x2.w);
        au[6] = pack_bf16(x3.x, x3.y);  au[7] = pack_bf16(x3.z, x3.w);
        v16bf afrag = __builtin_bit_cast(v16bf, au);

        #pragma unroll
        for (int j = 0; j < 8; ++j) {      // 8 N-tiles -> U = 128
            v16bf bfrag = frag_from_lds(&w1s[(j * 16 + lane16) * 32 + hi * 8]);
            acc[j] = __builtin_amdgcn_wmma_f32_16x16x32_bf16(
                         false, afrag, false, bfrag, (short)0, acc[j], false, false);
        }
    }

    // tanh (native V_TANH_F32), write hidden tile to LDS [m][u] for A-frag re-read
    #pragma unroll
    for (int j = 0; j < 8; ++j) {
        #pragma unroll
        for (int r = 0; r < 8; ++r) {
            int m = r + hi * 8;            // C layout: VGPR r -> M=r / r+8
            int u = j * 16 + lane16;       // N = lane16
            hids[wave][m * kU + u] = f2bf(fast_tanh(acc[j][r]));
        }
    }
    __syncthreads();

    // logits(16x16) = hid(16x128) @ W2(128x16), 4 WMMA K-steps
    v8f lacc = zero;
    #pragma unroll
    for (int ks = 0; ks < kU; ks += 32) {
        v16bf af  = frag_from_lds(&hids[wave][lane16 * kU + ks + hi * 8]);
        v16bf bf2 = frag_from_lds(&w2s[lane16 * kU + ks + hi * 8]);
        lacc = __builtin_amdgcn_wmma_f32_16x16x32_bf16(
                   false, af, false, bf2, (short)0, lacc, false, false);
    }

    // mask + store: element (m, a) with m = r + 8*hi, a = lane16
    #pragma unroll
    for (int r = 0; r < 8; ++r) {
        int m   = r + hi * 8;
        int row = rowTile0 + wave * 16 + m;
        float mv = mask[row];
        float lv = lacc[r] * mv + (1.0f - mv) * kNEG;
        logits[(size_t)row * kA + lane16] = lv;
    }
}

// ---------------------------------------------------------------------------
// K2: per-batch softmax over S for each of A heads, collapsed to scalar
//     wbar[b,s] = (1/A) * sum_a softmax_s(logits)[s,a]
// ---------------------------------------------------------------------------
__global__ __launch_bounds__(256) void k2_softmax_wbar(
    const float* __restrict__ logits, float* __restrict__ wbar)
{
    const int b    = blockIdx.x;
    const int tid  = threadIdx.x;
    const int a    = tid & 15;
    const int chnk = tid >> 4;            // 16 chunks of 256 s-values
    __shared__ float red[16][16];         // [a][chunk]
    __shared__ float maxa[16], rden[16];

    const float* L = logits + (size_t)b * kS * kA;

    float mx = -INFINITY;
    for (int s = chnk * 256; s < (chnk + 1) * 256; ++s)
        mx = fmaxf(mx, L[s * kA + a]);
    red[a][chnk] = mx;
    __syncthreads();
    if (tid < 16) {
        float m = red[tid][0];
        for (int c = 1; c < 16; ++c) m = fmaxf(m, red[tid][c]);
        maxa[tid] = m;
    }
    __syncthreads();

    float sm = 0.0f, mA = maxa[a];
    for (int s = chnk * 256; s < (chnk + 1) * 256; ++s)
        sm += __expf(L[s * kA + a] - mA);
    red[a][chnk] = sm;
    __syncthreads();
    if (tid < 16) {
        float t = 0.0f;
        for (int c = 0; c < 16; ++c) t += red[tid][c];
        rden[tid] = 1.0f / t;
    }
    __syncthreads();

    for (int s = tid; s < kS; s += 256) {
        float w = 0.0f;
        #pragma unroll
        for (int aa = 0; aa < 16; ++aa)
            w += __expf(L[s * kA + aa] - maxa[aa]) * rden[aa];
        wbar[b * kS + s] = w * (1.0f / 16.0f);
    }
}

__global__ void k_zero(float* __restrict__ p, int n) {
    int i = blockIdx.x * 256 + threadIdx.x;
    if (i < n) p[i] = 0.0f;
}

// ---------------------------------------------------------------------------
// K3: out[b,h] = sum_s wbar[b,s] * data[b,s,h]
// thread owns 4 consecutive h -> one global_load_b128 per row, coalesced.
// ---------------------------------------------------------------------------
__global__ __launch_bounds__(256) void k3_attend(
    const float* __restrict__ data, const float* __restrict__ wbar,
    float* __restrict__ out)
{
    const int b   = blockIdx.x;
    const int sc  = blockIdx.y;           // 16 S-chunks of 256
    const int tid = threadIdx.x;
    float4 acc = {0.f, 0.f, 0.f, 0.f};
    const int s0 = sc * 256;
    for (int s = s0; s < s0 + 256; ++s) {
        float w = wbar[b * kS + s];
        const float4* dp = (const float4*)(data + ((size_t)b * kS + s) * kH);
        float4 d = dp[tid];
        acc.x += w * d.x;  acc.y += w * d.y;
        acc.z += w * d.z;  acc.w += w * d.w;
    }
    float* o = &out[b * kH + tid * 4];
    atomicAdd(o + 0, acc.x);
    atomicAdd(o + 1, acc.y);
    atomicAdd(o + 2, acc.z);
    atomicAdd(o + 3, acc.w);
}

// ---------------------------------------------------------------------------
extern "C" void kernel_launch(void* const* d_in, const int* in_sizes, int n_in,
                              void* d_out, int out_size, void* d_ws, size_t ws_size,
                              hipStream_t stream)
{
    const float* data = (const float*)d_in[0];   // (16,4096,1024)
    const float* mask = (const float*)d_in[1];   // (16,4096)
    const float* W1   = (const float*)d_in[2];   // (1024,128)
    const float* W2   = (const float*)d_in[3];   // (128,16)
    float* out = (float*)d_out;                  // (16,1024)

    float*  logits = (float*)d_ws;                       // 16*4096*16 f32 = 4 MB
    float*  wbar   = logits + (size_t)kB * kS * kA;      // 16*4096 f32 = 256 KB
    __bf16* w1t    = (__bf16*)(wbar + (size_t)kB * kS);  // 1024*128 bf16 = 256 KB

    k0_w1pack      <<<dim3(kH * kU / 256), dim3(256), 0, stream>>>(W1, w1t);
    k1_logits      <<<dim3(kB * kS / 128), dim3(256), 0, stream>>>(data, mask, w1t, W2, logits);
    k2_softmax_wbar<<<dim3(kB),            dim3(256), 0, stream>>>(logits, wbar);
    k_zero         <<<dim3((kB * kH + 255) / 256), dim3(256), 0, stream>>>(out, kB * kH);
    k3_attend      <<<dim3(kB, 16),        dim3(256), 0, stream>>>(data, wbar, out);
}
